// BasicTransformerBlock_25331717112550
// MI455X (gfx1250) — matrix operand order repaired and checked by reading, not compiled, not run on hardware
//
#include <hip/hip_runtime.h>

typedef __attribute__((ext_vector_type(16))) _Float16 v16h;
typedef __attribute__((ext_vector_type(8)))  _Float16 v8h;
typedef __attribute__((ext_vector_type(8)))  float    v8f;

#define N_HEADS 16
#define HD      64
#define EDIM    1024
#define NT_HID  4096   // L*B tokens
#define NT_ENC  1024   // TEXT*B tokens
#define S_ATT   1024   // g + text
#define BH_ATT  128    // (SPARSE_N*B) * HEADS

// ---------------------------------------------------------------- fragments
__device__ __forceinline__ v8f wmma32(v16h a, v16h b, v8f c) {
  return __builtin_amdgcn_wmma_f32_16x16x32_f16(false, a, false, b, (short)0, c,
                                                false, false);
}

// A fragment (16x32 f16, M x K): lane L holds row m=L%16;
// elems 0..7 = K[h*8 .. h*8+7], elems 8..15 = K[16+h*8 .. 16+h*8+7], h=L>>4
__device__ __forceinline__ v16h frag_a(const _Float16* base, int ld) {
  int lane = threadIdx.x & 31;
  const _Float16* p = base + (size_t)(lane & 15) * ld + (lane >> 4) * 8;
  v8h lo = *(const v8h*)p;
  v8h hi = *(const v8h*)(p + 16);
  v16h f;
#pragma unroll
  for (int i = 0; i < 8; ++i) { f[i] = lo[i]; f[i + 8] = hi[i]; }
  return f;
}

// B fragment (32x16 f16, K x N) loaded from row-major B^T (N x K):
// lane L holds column n=L%16; elems 0..7 = K[h*8 .. h*8+7], elems 8..15 = K[16+h*8 .. 16+h*8+7], h=L>>4
__device__ __forceinline__ v16h frag_b(const _Float16* base, int ld) {
  int lane = threadIdx.x & 31;
  const _Float16* p = base + (size_t)(lane & 15) * ld + (lane >> 4) * 8;
  v8h lo = *(const v8h*)p;
  v8h hi = *(const v8h*)(p + 16);
  v16h f;
#pragma unroll
  for (int i = 0; i < 8; ++i) { f[i] = lo[i]; f[i + 8] = hi[i]; }
  return f;
}

// ---------------------------------------------------------------- f32 -> f16
__global__ void __launch_bounds__(256) cvt_f32_f16(const float* __restrict__ s,
                                                   _Float16* __restrict__ d, int n) {
  int i = blockIdx.x * 256 + threadIdx.x;
  if (i < n) d[i] = (_Float16)s[i];
}

// ---------------------------------------------------------------- GEMM
// C[M,N] = A[M,K](f16) * W[N,K]^T(f16) + bias[N], f32 accumulate/output.
// Block: 128 thr = 4 waves; wave = 32x64 tile; block = 128x64.
__global__ void __launch_bounds__(128) gemm_bias_f16(
    const _Float16* __restrict__ A, const _Float16* __restrict__ W,
    const float* __restrict__ bias, float* __restrict__ C, int M, int N, int K) {
  const int wv = threadIdx.x >> 5, lane = threadIdx.x & 31;
  const int n0 = blockIdx.x * 64;
  const int m0 = blockIdx.y * 128 + wv * 32;
  v8f acc[2][4];
#pragma unroll
  for (int i = 0; i < 2; ++i)
#pragma unroll
    for (int j = 0; j < 4; ++j)
#pragma unroll
      for (int r = 0; r < 8; ++r) acc[i][j][r] = 0.f;

  for (int k0 = 0; k0 < K; k0 += 32) {
    v16h a0 = frag_a(A + (size_t)m0 * K + k0, K);
    v16h a1 = frag_a(A + (size_t)(m0 + 16) * K + k0, K);
    v16h b0 = frag_b(W + (size_t)(n0 +  0) * K + k0, K);
    v16h b1 = frag_b(W + (size_t)(n0 + 16) * K + k0, K);
    v16h b2 = frag_b(W + (size_t)(n0 + 32) * K + k0, K);
    v16h b3 = frag_b(W + (size_t)(n0 + 48) * K + k0, K);
    acc[0][0] = wmma32(a0, b0, acc[0][0]);
    acc[0][1] = wmma32(a0, b1, acc[0][1]);
    acc[0][2] = wmma32(a0, b2, acc[0][2]);
    acc[0][3] = wmma32(a0, b3, acc[0][3]);
    acc[1][0] = wmma32(a1, b0, acc[1][0]);
    acc[1][1] = wmma32(a1, b1, acc[1][1]);
    acc[1][2] = wmma32(a1, b2, acc[1][2]);
    acc[1][3] = wmma32(a1, b3, acc[1][3]);
  }
  const int n = lane & 15, h = lane >> 4;
#pragma unroll
  for (int nb = 0; nb < 4; ++nb) {
    float bi = bias[n0 + nb * 16 + n];
#pragma unroll
    for (int mt = 0; mt < 2; ++mt)
#pragma unroll
      for (int r = 0; r < 8; ++r)
        C[(size_t)(m0 + mt * 16 + r + 8 * h) * N + n0 + nb * 16 + n] =
            acc[mt][nb][r] + bi;
  }
}

// ---------------------------------------------------------------- RoPE helper
__device__ __forceinline__ float rope_elem(int d, const float* xb, int pt, int py,
                                           int px, const float* ct, const float* st,
                                           const float* cy, const float* sy,
                                           const float* cx, const float* sx) {
  float v = xb[d], c, s, partner;
  if (d < 16) {            // t chunk, D=16
    c = ct[pt * 16 + d]; s = st[pt * 16 + d];
    partner = (d < 8) ? -xb[d + 8] : xb[d - 8];
  } else if (d < 40) {     // y chunk, D=24
    int dl = d - 16;
    c = cy[py * 24 + dl]; s = sy[py * 24 + dl];
    partner = (dl < 12) ? -xb[d + 12] : xb[d - 12];
  } else {                 // x chunk, D=24
    int dl = d - 40;
    c = cx[px * 24 + dl]; s = sx[px * 24 + dl];
    partner = (dl < 12) ? -xb[d + 12] : xb[d - 12];
  }
  return v * c + partner * s;
}

// ------------------------------------------------ pack hidden q/k: rms+rope
// one wave per (token t, head hh); writes Qh/Kh f16 [bh][1024][64]
__global__ void __launch_bounds__(256) pack_qk_hid(
    const float* __restrict__ Y, const float* __restrict__ nw,
    const float* __restrict__ ct, const float* __restrict__ st,
    const float* __restrict__ cy, const float* __restrict__ sy,
    const float* __restrict__ cx, const float* __restrict__ sx,
    const int* __restrict__ pt_a, const int* __restrict__ py_a,
    const int* __restrict__ px_a, _Float16* __restrict__ dst) {
  __shared__ float xb[8][64];
  int wv = threadIdx.x >> 5, lane = threadIdx.x & 31;
  int wid = blockIdx.x * 8 + wv;
  int t = wid >> 4, hh = wid & 15;
  const float* row = Y + (size_t)t * EDIM + hh * HD;
  float x0 = row[lane], x1 = row[lane + 32];
  float ss = x0 * x0 + x1 * x1;
#pragma unroll
  for (int off = 16; off; off >>= 1) ss += __shfl_xor(ss, off, 32);
  float rn = rsqrtf(ss * (1.f / 64.f) + 1e-6f);
  x0 = x0 * rn * nw[lane]; x1 = x1 * rn * nw[lane + 32];
  xb[wv][lane] = x0; xb[wv][lane + 32] = x1;
  __syncthreads();
  int pt = pt_a[t], py = py_a[t], px = px_a[t];
  float o0 = rope_elem(lane,      xb[wv], pt, py, px, ct, st, cy, sy, cx, sx);
  float o1 = rope_elem(lane + 32, xb[wv], pt, py, px, ct, st, cy, sy, cx, sx);
  int l = t >> 1, b = t & 1;
  int s = l >> 2, bp = ((l & 3) << 1) | b;
  _Float16* drow = dst + ((size_t)(bp * N_HEADS + hh) * S_ATT + s) * HD;
  drow[lane] = (_Float16)o0; drow[lane + 32] = (_Float16)o1;
}

// ------------------------------------------------ pack encoder q/k: rms, x4
__global__ void __launch_bounds__(256) pack_qk_enc(
    const float* __restrict__ Y, const float* __restrict__ nw,
    _Float16* __restrict__ dst) {
  int wv = threadIdx.x >> 5, lane = threadIdx.x & 31;
  int wid = blockIdx.x * 8 + wv;
  int t = wid >> 4, hh = wid & 15;     // t = ts*2 + b
  const float* row = Y + (size_t)t * EDIM + hh * HD;
  float x0 = row[lane], x1 = row[lane + 32];
  float ss = x0 * x0 + x1 * x1;
#pragma unroll
  for (int off = 16; off; off >>= 1) ss += __shfl_xor(ss, off, 32);
  float rn = rsqrtf(ss * (1.f / 64.f) + 1e-6f);
  _Float16 h0 = (_Float16)(x0 * rn * nw[lane]);
  _Float16 h1 = (_Float16)(x1 * rn * nw[lane + 32]);
  int ts = t >> 1, b = t & 1, s = 512 + ts;
#pragma unroll
  for (int k = 0; k < 4; ++k) {
    int bh = (k * 2 + b) * N_HEADS + hh;
    _Float16* drow = dst + ((size_t)bh * S_ATT + s) * HD;
    drow[lane] = h0; drow[lane + 32] = h1;
  }
}

// ------------------------------------------------ V transpose pack (LDS tile)
// Vt f16 [bh][64 d][1024 s].  grid (128 bh, 8 s-tiles), enc selects half.
__global__ void __launch_bounds__(256) vt_pack(const float* __restrict__ Y,
                                               _Float16* __restrict__ Vt, int enc) {
  __shared__ float tile[64][65];
  int bh = blockIdx.x, st = blockIdx.y;
  int bp = bh >> 4, hh = bh & 15;
  int s_base = enc ? (512 + st * 64) : (st * 64);
  for (int i = threadIdx.x; i < 64 * 64; i += 256) {
    int si = i >> 6, d = i & 63;
    int s = s_base + si;
    int t = enc ? ((s - 512) * 2 + (bp & 1)) : (s * 8 + bp);
    tile[si][d] = Y[(size_t)t * EDIM + hh * HD + d];
  }
  __syncthreads();
  for (int i = threadIdx.x; i < 64 * 64; i += 256) {
    int d = i >> 6, si = i & 63;
    Vt[((size_t)bh * HD + d) * S_ATT + s_base + si] = (_Float16)tile[si][d];
  }
}

// ------------------------------------------------ flash attention, fixed-shift
// Q,K rows are RMS-normalized => |score| <= 8 (Cauchy-Schwarz), so softmax can
// use a constant shift exp(s-8): no online max, no rescales, deferred row-sum.
// grid (128 bh, 16); 4 waves/block; wave = one 16-row q tile, 16 KV steps of 64.
__global__ void __launch_bounds__(128) attn_flash(const _Float16* __restrict__ Qh,
                                                  const _Float16* __restrict__ Kh,
                                                  const _Float16* __restrict__ Vt,
                                                  _Float16* __restrict__ O) {
  __shared__ __align__(16) _Float16 P[4][16][72];   // 64 cols + 8 pad
  int bh = blockIdx.x;
  int wv = threadIdx.x >> 5, lane = threadIdx.x & 31;
  int qt = blockIdx.y * 4 + wv;
  int n = lane & 15, h = lane >> 4;

  const _Float16* Qb = Qh + ((size_t)bh * S_ATT + qt * 16) * HD;
  v16h qa0 = frag_a(Qb, HD);
  v16h qa1 = frag_a(Qb + 32, HD);
  qa0 = qa0 * (_Float16)0.125f;   // fold 1/sqrt(hd) into Q (exact, pow2)
  qa1 = qa1 * (_Float16)0.125f;

  float lsum[8];
  v8f acc[4];
#pragma unroll
  for (int r = 0; r < 8; ++r) lsum[r] = 0.f;
#pragma unroll
  for (int nb = 0; nb < 4; ++nb)
#pragma unroll
    for (int r = 0; r < 8; ++r) acc[nb][r] = 0.f;

  for (int j0 = 0; j0 < S_ATT; j0 += 64) {
    const _Float16* Kb = Kh + ((size_t)bh * S_ATT + j0) * HD;
    v8f s[4];
#pragma unroll
    for (int nb = 0; nb < 4; ++nb) {
      v16h kb0 = frag_b(Kb + nb * 16 * HD, HD);
      v16h kb1 = frag_b(Kb + nb * 16 * HD + 32, HD);
      v8f z;
#pragma unroll
      for (int r = 0; r < 8; ++r) z[r] = 0.f;
      z = wmma32(qa0, kb0, z);
      s[nb] = wmma32(qa1, kb1, z);
    }
    // p = exp(s - 8); accumulate per-lane partial row sums (reduced after loop)
#pragma unroll
    for (int nb = 0; nb < 4; ++nb)
#pragma unroll
      for (int r = 0; r < 8; ++r) {
        float p = __expf(s[nb][r] - 8.0f);
        s[nb][r] = p;
        lsum[r] += p;
      }
    // C-layout -> A-layout via per-wave LDS tile
#pragma unroll
    for (int nb = 0; nb < 4; ++nb)
#pragma unroll
      for (int r = 0; r < 8; ++r)
        P[wv][r + 8 * h][nb * 16 + n] = (_Float16)s[nb][r];
    __syncthreads();
    v16h pa0 = frag_a(&P[wv][0][0], 72);
    v16h pa1 = frag_a(&P[wv][0][0] + 32, 72);
    const _Float16* Vb = Vt + (size_t)bh * HD * S_ATT + j0;
#pragma unroll
    for (int nb = 0; nb < 4; ++nb) {
      v16h vb0 = frag_b(Vb + (size_t)(nb * 16) * S_ATT, S_ATT);
      v16h vb1 = frag_b(Vb + (size_t)(nb * 16) * S_ATT + 32, S_ATT);
      acc[nb] = wmma32(pa0, vb0, acc[nb]);
      acc[nb] = wmma32(pa1, vb1, acc[nb]);
    }
    __syncthreads();
  }
  // single deferred row-sum reduction across the 16-lane half
#pragma unroll
  for (int r = 0; r < 8; ++r) {
    float t = lsum[r];
    t += __shfl_xor(t, 1, 16);
    t += __shfl_xor(t, 2, 16);
    t += __shfl_xor(t, 4, 16);
    t += __shfl_xor(t, 8, 16);
    lsum[r] = 1.f / t;
  }
  int bp = bh >> 4, hh = bh & 15;
#pragma unroll
  for (int nb = 0; nb < 4; ++nb)
#pragma unroll
    for (int r = 0; r < 8; ++r) {
      int s = qt * 16 + r + 8 * h;
      O[(size_t)(s * 8 + bp) * EDIM + hh * HD + nb * 16 + n] =
          (_Float16)(acc[nb][r] * lsum[r]);
    }
}

// ------------------------------------------------ mean over 4 sparse replicas
__global__ void __launch_bounds__(256) mean_enc(const _Float16* __restrict__ O,
                                                _Float16* __restrict__ Om) {
  int i = blockIdx.x * 256 + threadIdx.x;   // over 1024*1024
  int t2 = i >> 10, c = i & 1023;
  int ts = t2 >> 1, b = t2 & 1;
  float s = 0.f;
#pragma unroll
  for (int k = 0; k < 4; ++k)
    s += (float)O[(size_t)(NT_HID + ts * 8 + k * 2 + b) * EDIM + c];
  Om[i] = (_Float16)(s * 0.25f);
}

// ---------------------------------------------------------------- launch
extern "C" void kernel_launch(void* const* d_in, const int* in_sizes, int n_in,
                              void* d_out, int out_size, void* d_ws, size_t ws_size,
                              hipStream_t stream) {
  const float* hidden  = (const float*)d_in[0];
  const float* encoder = (const float*)d_in[1];
  const float* Wq = (const float*)d_in[2];  const float* bq = (const float*)d_in[3];
  const float* Wk = (const float*)d_in[4];  const float* bk = (const float*)d_in[5];
  const float* Wv = (const float*)d_in[6];  const float* bv = (const float*)d_in[7];
  const float* Waq = (const float*)d_in[8];  const float* baq = (const float*)d_in[9];
  const float* Wak = (const float*)d_in[10]; const float* bak = (const float*)d_in[11];
  const float* Wav = (const float*)d_in[12]; const float* bav = (const float*)d_in[13];
  const float* nq_w = (const float*)d_in[14]; const float* nk_w = (const float*)d_in[15];
  const float* naq_w = (const float*)d_in[16]; const float* nak_w = (const float*)d_in[17];
  const float* Wout = (const float*)d_in[18]; const float* bout = (const float*)d_in[19];
  const float* Wadd = (const float*)d_in[20]; const float* badd = (const float*)d_in[21];
  const float* cos_t = (const float*)d_in[22]; const float* sin_t = (const float*)d_in[23];
  const float* cos_y = (const float*)d_in[24]; const float* sin_y = (const float*)d_in[25];
  const float* cos_x = (const float*)d_in[26]; const float* sin_x = (const float*)d_in[27];
  const int* pos_t = (const int*)d_in[28];
  const int* pos_y = (const int*)d_in[29];
  const int* pos_x = (const int*)d_in[30];

  const size_t MB = 1024u * 1024u;
  char* ws = (char*)d_ws;
  _Float16* Xh   = (_Float16*)(ws + 0);        //  8 MB : hidden tokens f16
  _Float16* Xe   = (_Float16*)(ws + 8 * MB);   //  2 MB : encoder tokens f16
  _Float16* Wcur = (_Float16*)(ws + 10 * MB);  //  2 MB : current weight f16
  float*    Yb   = (float*)   (ws + 12 * MB);  // 16 MB : gemm f32 scratch
  _Float16* Qh   = (_Float16*)(ws + 28 * MB);  // 16 MB : [128][1024][64]
  _Float16* Kh   = (_Float16*)(ws + 44 * MB);  // 16 MB
  _Float16* Vt   = (_Float16*)(ws + 60 * MB);  // 16 MB : [128][64][1024]
  _Float16* Oatt = (_Float16*)(ws + 76 * MB);  // 16 MB : [8192][1024]
  _Float16* Om   = (_Float16*)(ws + 92 * MB);  //  2 MB : enc-mean tokens f16
  float* out = (float*)d_out;

  auto cvt = [&](const float* s, _Float16* d, int n) {
    cvt_f32_f16<<<(n + 255) / 256, 256, 0, stream>>>(s, d, n);
  };
  dim3 gHid(EDIM / 64, NT_HID / 128);  // (16,32)
  dim3 gEnc(EDIM / 64, NT_ENC / 128);  // (16, 8)

  cvt(hidden, Xh, NT_HID * EDIM);
  cvt(encoder, Xe, NT_ENC * EDIM);

  // ---- Q (hidden): gemm -> rms+rope pack
  cvt(Wq, Wcur, EDIM * EDIM);
  gemm_bias_f16<<<gHid, 128, 0, stream>>>(Xh, Wcur, bq, Yb, NT_HID, EDIM, EDIM);
  pack_qk_hid<<<NT_HID * N_HEADS / 8, 256, 0, stream>>>(
      Yb, nq_w, cos_t, sin_t, cos_y, sin_y, cos_x, sin_x, pos_t, pos_y, pos_x, Qh);
  // ---- K (hidden)
  cvt(Wk, Wcur, EDIM * EDIM);
  gemm_bias_f16<<<gHid, 128, 0, stream>>>(Xh, Wcur, bk, Yb, NT_HID, EDIM, EDIM);
  pack_qk_hid<<<NT_HID * N_HEADS / 8, 256, 0, stream>>>(
      Yb, nk_w, cos_t, sin_t, cos_y, sin_y, cos_x, sin_x, pos_t, pos_y, pos_x, Kh);
  // ---- V (hidden) -> transposed layout
  cvt(Wv, Wcur, EDIM * EDIM);
  gemm_bias_f16<<<gHid, 128, 0, stream>>>(Xh, Wcur, bv, Yb, NT_HID, EDIM, EDIM);
  vt_pack<<<dim3(BH_ATT, 8), 256, 0, stream>>>(Yb, Vt, 0);
  // ---- encoder Q
  cvt(Waq, Wcur, EDIM * EDIM);
  gemm_bias_f16<<<gEnc, 128, 0, stream>>>(Xe, Wcur, baq, Yb, NT_ENC, EDIM, EDIM);
  pack_qk_enc<<<NT_ENC * N_HEADS / 8, 256, 0, stream>>>(Yb, naq_w, Qh);
  // ---- encoder K
  cvt(Wak, Wcur, EDIM * EDIM);
  gemm_bias_f16<<<gEnc, 128, 0, stream>>>(Xe, Wcur, bak, Yb, NT_ENC, EDIM, EDIM);
  pack_qk_enc<<<NT_ENC * N_HEADS / 8, 256, 0, stream>>>(Yb, nak_w, Kh);
  // ---- encoder V
  cvt(Wav, Wcur, EDIM * EDIM);
  gemm_bias_f16<<<gEnc, 128, 0, stream>>>(Xe, Wcur, bav, Yb, NT_ENC, EDIM, EDIM);
  vt_pack<<<dim3(BH_ATT, 8), 256, 0, stream>>>(Yb, Vt, 1);

  // ---- fused flash attention (128 bh x 64 q-tiles)
  attn_flash<<<dim3(BH_ATT, 16), 128, 0, stream>>>(Qh, Kh, Vt, Oatt);

  // ---- encoder replica mean
  mean_enc<<<NT_ENC * EDIM / 256, 256, 0, stream>>>(Oatt, Om);

  // ---- output projections straight into d_out (f32)
  cvt(Wout, Wcur, EDIM * EDIM);
  gemm_bias_f16<<<gHid, 128, 0, stream>>>(Oatt, Wcur, bout, out, NT_HID, EDIM, EDIM);
  cvt(Wadd, Wcur, EDIM * EDIM);
  gemm_bias_f16<<<gEnc, 128, 0, stream>>>(Om, Wcur, badd, out + (size_t)NT_HID * EDIM,
                                          NT_ENC, EDIM, EDIM);
}
